// GaussianMRIModel_10866267259092
// MI455X (gfx1250) — matrix operand order repaired and checked by reading, not compiled, not run on hardware
//
#include <hip/hip_runtime.h>

typedef __attribute__((ext_vector_type(16))) _Float16 v16h;
typedef __attribute__((ext_vector_type(8)))  float    v8f;

#define NZV 64
#define NXV 192
#define NYV 192
#define TX 8
#define TY 16
#define CUT2 9.0f
#define EPSV 1e-8f
#define GPASS 256   // gaussians culled per pass (== blockDim.x)

// Gather-style voxelizer: each block owns a 1 x TX x TY voxel tile.
// Per pass of 256 gaussians: bbox-cull into a compacted LDS list, then each
// wave accumulates its 16-voxel column group with v_wmma_f32_16x16x32_f16:
//   D(16x16,f32) += A(rho 16x32,f16) * B(w 32x16,f16)
// Row 0 of A = rho_real, row 1 = rho_imag -> C vgpr0/vgpr1 of lanes 0..15
// hold (vr, vi) for voxel column n, written as one coalesced float2.
__global__ __launch_bounds__(256) void gaussian_voxelize_wmma(
    const float* __restrict__ centers,     // (M,3): z,x,y
    const float* __restrict__ log_scales,  // (M,3): z,x,y
    const float* __restrict__ rho_real,    // (M,)
    const float* __restrict__ rho_imag,    // (M,)
    float* __restrict__ out,               // (NZ,NX,NY) complex64 -> interleaved f32
    int M)
{
    __shared__ float spz[GPASS], spx[GPASS], spy[GPASS];
    __shared__ float siz[GPASS], six[GPASS], siy[GPASS];
    __shared__ float srr[GPASS], sri[GPASS];
    __shared__ int   scount;

    const int tid  = threadIdx.x;
    const int lane = tid & 31;
    const int wv   = tid >> 5;      // wave id 0..7 -> x offset in tile
    const int n    = lane & 15;     // voxel column / A-row index
    const int half = lane >> 4;     // K-half selector for A and B layouts

    const int x0 = blockIdx.x * TX;
    const int y0 = blockIdx.y * TY;
    const int z  = blockIdx.z;

    // voxel coordinates for this lane's column (z plane, x row = wave, y = n)
    const float czv = -1.0f + 2.0f * (float)z          / (float)(NZV - 1);
    const float cxv = -1.0f + 2.0f * (float)(x0 + wv)  / (float)(NXV - 1);
    const float cyv = -1.0f + 2.0f * (float)(y0 + n)   / (float)(NYV - 1);

    // tile bounds for culling (tile spans a single z-plane: loZ == hiZ == czv)
    const float loZ = czv;
    const float hiZ = czv;
    const float loX = -1.0f + 2.0f * (float)x0            / (float)(NXV - 1);
    const float hiX = -1.0f + 2.0f * (float)(x0 + TX - 1) / (float)(NXV - 1);
    const float loY = -1.0f + 2.0f * (float)y0            / (float)(NYV - 1);
    const float hiY = -1.0f + 2.0f * (float)(y0 + TY - 1) / (float)(NYV - 1);

    v8f acc = {};  // fp32 accumulator tile (persistent across all passes)

    const int npasses = (M + GPASS - 1) / GPASS;
    for (int p = 0; p < npasses; ++p) {
        __syncthreads();  // previous pass fully consumed
        // zero slot tid -> padding entries give w=1 * rho=0 = 0 contribution
        spz[tid] = 0.0f; spx[tid] = 0.0f; spy[tid] = 0.0f;
        siz[tid] = 0.0f; six[tid] = 0.0f; siy[tid] = 0.0f;
        srr[tid] = 0.0f; sri[tid] = 0.0f;
        if (tid == 0) scount = 0;
        __syncthreads();

        const int g = p * GPASS + tid;
        if (g + GPASS < M)
            __builtin_prefetch(centers + 3 * (g + GPASS), 0, 0);  // global_prefetch_b8
        if (g < M) {
            const float pz = centers[3 * g + 0];
            const float px = centers[3 * g + 1];
            const float py = centers[3 * g + 2];
            const float sz = __expf(log_scales[3 * g + 0]) + EPSV;
            const float sx = __expf(log_scales[3 * g + 1]) + EPSV;
            const float sy = __expf(log_scales[3 * g + 2]) + EPSV;
            const bool hit =
                (pz >= loZ - 3.0f * sz) && (pz <= hiZ + 3.0f * sz) &&
                (px >= loX - 3.0f * sx) && (px <= hiX + 3.0f * sx) &&
                (py >= loY - 3.0f * sy) && (py <= hiY + 3.0f * sy);
            if (hit) {
                const int slot = atomicAdd(&scount, 1);
                spz[slot] = pz; spx[slot] = px; spy[slot] = py;
                siz[slot] = 1.0f / sz; six[slot] = 1.0f / sx; siy[slot] = 1.0f / sy;
                srr[slot] = rho_real[g];
                sri[slot] = rho_imag[g];
            }
        }
        __syncthreads();

        const int cnt = scount;                 // uniform across block
        const int nchunks = (cnt + 31) >> 5;
        for (int ch = 0; ch < nchunks; ++ch) {
            const int kb = ch << 5;

            // ---- B matrix (32x16 f16): w values for this lane's column ----
            // lane half 0 holds K=0..15, half 1 holds K=16..31; 2 Ks per VGPR.
            v16h b;
#pragma unroll
            for (int j = 0; j < 8; ++j) {
#pragma unroll
                for (int t = 0; t < 2; ++t) {
                    const int k = kb + half * 16 + 2 * j + t;
                    const float ddz = (czv - spz[k]) * siz[k];
                    const float ddx = (cxv - spx[k]) * six[k];
                    const float ddy = (cyv - spy[k]) * siy[k];
                    const float d2  = ddz * ddz + ddx * ddx + ddy * ddy;
                    const float w   = (d2 <= CUT2) ? __expf(-0.5f * d2) : 0.0f;
                    b[2 * j + t] = (_Float16)w;
                }
            }

            // ---- A matrix (16x32 f16): row 0 = rho_real, row 1 = rho_imag ----
            // lane half 0: vgpr0-7 -> K 0..15
            // lane half 1: vgpr0-7 -> K 16..31
            const float* rsel   = (n == 1) ? sri : srr;
            const float rowmask = (n < 2) ? 1.0f : 0.0f;
            v16h a;
#pragma unroll
            for (int j = 0; j < 8; ++j) {
                const int kA = half * 16 + 2 * j;
                a[2 * j + 0] = (_Float16)(rsel[kb + kA + 0] * rowmask);
                a[2 * j + 1] = (_Float16)(rsel[kb + kA + 1] * rowmask);
            }

            acc = __builtin_amdgcn_wmma_f32_16x16x32_f16(
                false, a, false, b, (short)0, acc, false, false);
        }
    }

    // D layout: lanes 0..15, vgpr0 = row M=0 (vr), vgpr1 = row M=1 (vi),
    // column N = lane. Lanes 16..31 hold rows 8..15 (all zero) -> skip.
    if (lane < 16) {
        const int idx = (z * NXV + (x0 + wv)) * NYV + (y0 + n);
        float2* o = (float2*)out;
        o[idx] = make_float2(acc[0], acc[1]);
    }
}

extern "C" void kernel_launch(void* const* d_in, const int* in_sizes, int n_in,
                              void* d_out, int out_size, void* d_ws, size_t ws_size,
                              hipStream_t stream) {
    const float* centers    = (const float*)d_in[0];
    const float* log_scales = (const float*)d_in[1];
    const float* rho_real   = (const float*)d_in[2];
    const float* rho_imag   = (const float*)d_in[3];
    float* out = (float*)d_out;
    const int M = in_sizes[2];  // rho_real element count

    dim3 grid(NXV / TX, NYV / TY, NZV);  // 24 x 12 x 64 tiles
    gaussian_voxelize_wmma<<<grid, 256, 0, stream>>>(
        centers, log_scales, rho_real, rho_imag, out, M);
}
